// Model_43619687858922
// MI455X (gfx1250) — matrix operand order repaired and checked by reading, not compiled, not run on hardware
//
#include <hip/hip_runtime.h>
#include <hip/hip_bf16.h>
#include <math.h>

// ---------------- problem constants ----------------
#define BB   8192
#define INN  512
#define HH1  256
#define HH2  64
#define DD   32
#define KC   256          // codebook entries
#define HH3  64
#define HH4  512
#define DK   (DD * HH2)   // 2048 = decoder input width
#define M2   (BB * HH2)   // 524288 VQ rows

typedef __bf16 bf16_t;
typedef __attribute__((ext_vector_type(16))) __bf16 v16bf;
typedef __attribute__((ext_vector_type(8)))  __bf16 v8bf;
typedef __attribute__((ext_vector_type(8)))  float  v8f;

// ---- WMMA fragment loaders (CDNA5 16-bit layouts, CDNA5 ISA §7.12.2) ----
// A (16x32, MxK): lane m=lane&15, half=lane>>4; element j: K = j + 8*half (+8 if j>=8)
//   -> two contiguous 16B chunks at K offsets 8*half and 8*half+16.
__device__ __forceinline__ v16bf load_frag_a(const bf16_t* __restrict__ row, int half) {
  union { v16bf v; v8bf h[2]; } u;
  u.h[0] = *(const v8bf*)(row + 8 * half);
  u.h[1] = *(const v8bf*)(row + 8 * half + 16);
  return u.v;
}
// B (32x16, KxN) fed from W^T rows ([N,K] storage): lane n=lane&15; element j: K = j + 8*half (+8 if j>=8)
//   -> two contiguous 16B chunks at K offsets 8*half and 8*half+16.
__device__ __forceinline__ v16bf load_frag_b(const bf16_t* __restrict__ row, int half) {
  union { v16bf v; v8bf h[2]; } u;
  u.h[0] = *(const v8bf*)(row + 8 * half);
  u.h[1] = *(const v8bf*)(row + 8 * half + 16);
  return u.v;
}

// ---------------- conversion / prep kernels ----------------
__global__ void cvt_bf16_kernel(const float* __restrict__ in, bf16_t* __restrict__ out, int n) {
  int i = blockIdx.x * 256 + threadIdx.x;
  if (i < n) out[i] = (bf16_t)in[i];
}

// in: [Kd, Nd] row-major fp32  ->  out: [Nd, Kd] row-major bf16 (W^T)
__global__ void transpose_bf16_kernel(const float* __restrict__ in, bf16_t* __restrict__ out,
                                      int Kd, int Nd) {
  int i = blockIdx.x * 256 + threadIdx.x;
  if (i >= Kd * Nd) return;
  int k = i / Nd, n = i - k * Nd;
  out[(size_t)n * Kd + k] = (bf16_t)in[i];
}

__global__ void codebook_prep_kernel(const float* __restrict__ cb, bf16_t* __restrict__ cbb,
                                     float* __restrict__ cnorm) {
  int k = threadIdx.x;           // 256 threads, one per code
  float s = 0.f;
  #pragma unroll
  for (int d = 0; d < DD; ++d) {
    float v = cb[k * DD + d];
    s += v * v;
    cbb[k * DD + d] = (bf16_t)v;
  }
  cnorm[k] = s;
}

__global__ void zero_hist_kernel(int* __restrict__ hist) { hist[threadIdx.x] = 0; }

// ---------------- generic bf16 WMMA GEMM: out = act(A[M,K] @ Bt[N,K]^T + bias) ----------------
__global__ __launch_bounds__(256)
void gemm_bf16_kernel(const bf16_t* __restrict__ A, const bf16_t* __restrict__ Bt,
                      const float* __restrict__ bias,
                      float* __restrict__ outF, bf16_t* __restrict__ outB,
                      int M, int N, int Kd, int relu) {
  const int wave = (blockIdx.x * 256 + threadIdx.x) >> 5;
  const int lane = threadIdx.x & 31;
  const int tilesN = N >> 5, tilesM = M >> 5;
  const int tm = wave / tilesN, tn = wave - tm * tilesN;
  if (tm >= tilesM) return;                 // wave-uniform: EXEC stays all-ones
  const int half = lane >> 4, l16 = lane & 15;

  const bf16_t* arow0 = A + (size_t)(tm * 32 + l16) * Kd;
  const bf16_t* arow1 = arow0 + (size_t)16 * Kd;
  const bf16_t* brow0 = Bt + (size_t)(tn * 32 + l16) * Kd;
  const bf16_t* brow1 = brow0 + (size_t)16 * Kd;

  v8f c00 = {}, c01 = {}, c10 = {}, c11 = {};
  for (int k0 = 0; k0 < Kd; k0 += 32) {
    v16bf a0 = load_frag_a(arow0 + k0, half);
    v16bf a1 = load_frag_a(arow1 + k0, half);
    v16bf b0 = load_frag_b(brow0 + k0, half);
    v16bf b1 = load_frag_b(brow1 + k0, half);
    c00 = __builtin_amdgcn_wmma_f32_16x16x32_bf16(false, a0, false, b0, (short)0, c00, false, false);
    c01 = __builtin_amdgcn_wmma_f32_16x16x32_bf16(false, a0, false, b1, (short)0, c01, false, false);
    c10 = __builtin_amdgcn_wmma_f32_16x16x32_bf16(false, a1, false, b0, (short)0, c10, false, false);
    c11 = __builtin_amdgcn_wmma_f32_16x16x32_bf16(false, a1, false, b1, (short)0, c11, false, false);
  }

  const int n0 = tn * 32 + l16;
  const float bia0 = bias ? bias[n0] : 0.f;
  const float bia1 = bias ? bias[n0 + 16] : 0.f;
  #pragma unroll
  for (int r = 0; r < 8; ++r) {
    int m0 = tm * 32 + r + 8 * half;        // C/D layout: M = r + 8*half, N = lane&15
    float v00 = c00[r] + bia0, v01 = c01[r] + bia1;
    float v10 = c10[r] + bia0, v11 = c11[r] + bia1;
    if (relu) {
      v00 = fmaxf(v00, 0.f); v01 = fmaxf(v01, 0.f);
      v10 = fmaxf(v10, 0.f); v11 = fmaxf(v11, 0.f);
    }
    size_t o00 = (size_t)m0 * N + n0;
    size_t o10 = o00 + (size_t)16 * N;
    if (outF) { outF[o00] = v00; outF[o00 + 16] = v01; outF[o10] = v10; outF[o10 + 16] = v11; }
    if (outB) { outB[o00] = (bf16_t)v00; outB[o00 + 16] = (bf16_t)v01;
                outB[o10] = (bf16_t)v10; outB[o10 + 16] = (bf16_t)v11; }
  }
}

// ---------------- pre-VQ conv (1->32 channels, k=3, SAME), bf16 out [M2, D] ----------------
__global__ void conv_prevq_kernel(const float* __restrict__ z, const float* __restrict__ cw,
                                  const float* __restrict__ cbias, bf16_t* __restrict__ flatb) {
  size_t gid = (size_t)blockIdx.x * 256 + threadIdx.x;     // over M2*D
  if (gid >= (size_t)M2 * DD) return;
  int d = (int)(gid & (DD - 1));
  size_t r = gid >> 5;
  int t = (int)(r & (HH2 - 1));
  size_t b = r >> 6;
  const float* zr = z + b * HH2;
  float zm = (t > 0)       ? zr[t - 1] : 0.f;
  float z0 = zr[t];
  float zp = (t < HH2 - 1) ? zr[t + 1] : 0.f;
  float v = cbias[d] + cw[d * 3 + 0] * zm + cw[d * 3 + 1] * z0 + cw[d * 3 + 2] * zp;
  flatb[gid] = (bf16_t)v;
}

// ---------------- VQ argmin: one wave per 16 rows; one WMMA per 16-code tile ----------------
__global__ __launch_bounds__(256)
void vq_argmin_kernel(const bf16_t* __restrict__ flatb, const bf16_t* __restrict__ cbb,
                      const float* __restrict__ cnorm, int* __restrict__ idxOut) {
  const int wave = (blockIdx.x * 256 + threadIdx.x) >> 5;
  const int lane = threadIdx.x & 31;
  const int half = lane >> 4, l16 = lane & 15;
  const int rowBase = wave * 16;
  if (rowBase >= M2) return;                // wave-uniform

  v16bf a = load_frag_a(flatb + (size_t)(rowBase + l16) * DD, half);
  float minv[8]; int mini[8];
  #pragma unroll
  for (int r = 0; r < 8; ++r) { minv[r] = 3.4e38f; mini[r] = 0; }

  for (int n0 = 0; n0 < KC; n0 += 16) {
    v16bf b = load_frag_b(cbb + (size_t)(n0 + l16) * DD, half);
    v8f c = {};
    c = __builtin_amdgcn_wmma_f32_16x16x32_bf16(false, a, false, b, (short)0, c, false, false);
    float cn = cnorm[n0 + l16];
    int nid = n0 + l16;
    #pragma unroll
    for (int r = 0; r < 8; ++r) {
      float dd = cn - 2.0f * c[r];          // ||x||^2 term is argmin-invariant
      if (dd < minv[r]) { minv[r] = dd; mini[r] = nid; }
    }
  }
  // butterfly argmin over the 16 N-lanes of each half-wave
  #pragma unroll
  for (int m = 1; m <= 8; m <<= 1) {
    #pragma unroll
    for (int r = 0; r < 8; ++r) {
      float ov = __shfl_xor(minv[r], m, 32);
      int   oi = __shfl_xor(mini[r], m, 32);
      if (ov < minv[r] || (ov == minv[r] && oi < mini[r])) { minv[r] = ov; mini[r] = oi; }
    }
  }
  if (l16 == 0) {
    #pragma unroll
    for (int r = 0; r < 8; ++r) idxOut[rowBase + r + 8 * half] = mini[r];
  }
}

// ---------------- quantize: gather codes, MSE partials, histogram, outputs ----------------
__global__ __launch_bounds__(256)
void quantize_kernel(const int* __restrict__ idx, const float* __restrict__ cbf,
                     const float* __restrict__ z, const float* __restrict__ cw,
                     const float* __restrict__ cbias,
                     float* __restrict__ outQ, bf16_t* __restrict__ qb,
                     float* __restrict__ partials, int* __restrict__ hist) {
  __shared__ float wsum[8];
  const int wid = threadIdx.x >> 5, lane = threadIdx.x & 31;
  const size_t r = (size_t)blockIdx.x * 8 + wid;            // M2/8 blocks exactly
  const int t = (int)(r & (HH2 - 1));
  const size_t b = r >> 6;
  const int d = lane;
  const int code = idx[r];
  const float q = cbf[code * DD + d];
  // recompute z_t in fp32 (3 FMA) for an accurate MSE
  const float* zr = z + b * HH2;
  float zm = (t > 0)       ? zr[t - 1] : 0.f;
  float z0 = zr[t];
  float zp = (t < HH2 - 1) ? zr[t + 1] : 0.f;
  float zt = cbias[d] + cw[d * 3 + 0] * zm + cw[d * 3 + 1] * z0 + cw[d * 3 + 2] * zp;
  float df = q - zt;
  float s = df * df;
  #pragma unroll
  for (int m = 16; m >= 1; m >>= 1) s += __shfl_xor(s, m, 32);
  const size_t o = b * DK + (size_t)d * HH2 + t;            // [B, D, H2]
  outQ[o] = q;                                              // straight-through fwd == q
  qb[o]   = (bf16_t)q;                                      // decoder GEMM input
  if (lane == 0) { wsum[wid] = s; atomicAdd(&hist[code], 1); }
  __syncthreads();
  if (threadIdx.x == 0) {
    float tot = 0.f;
    #pragma unroll
    for (int i = 0; i < 8; ++i) tot += wsum[i];
    partials[blockIdx.x] = tot;
  }
}

// ---------------- finalize: loss + perplexity ----------------
__global__ void finalize_kernel(const float* __restrict__ partials, int npart,
                                const int* __restrict__ hist,
                                float* __restrict__ loss_out, float* __restrict__ perp_out) {
  __shared__ float red[256];
  float s = 0.f;
  for (int i = threadIdx.x; i < npart; i += 256) s += partials[i];
  red[threadIdx.x] = s; __syncthreads();
  for (int st = 128; st > 0; st >>= 1) {
    if (threadIdx.x < st) red[threadIdx.x] += red[threadIdx.x + st];
    __syncthreads();
  }
  float mse_sum = red[0];
  __syncthreads();
  float p = (float)hist[threadIdx.x] * (1.0f / (float)M2);
  red[threadIdx.x] = p * logf(p + 1e-10f);
  __syncthreads();
  for (int st = 128; st > 0; st >>= 1) {
    if (threadIdx.x < st) red[threadIdx.x] += red[threadIdx.x + st];
    __syncthreads();
  }
  if (threadIdx.x == 0) {
    float mse = mse_sum / (float)((size_t)M2 * DD);
    *loss_out = (1.0f + 0.25f) * mse;       // fwd: q_latent == e_latent == mse
    *perp_out = expf(-red[0]);
  }
}

// ---------------- host launcher ----------------
static inline int gemm_blocks(int M, int N) {
  int waves = (M >> 5) * (N >> 5);
  return (waves + 7) / 8;
}

extern "C" void kernel_launch(void* const* d_in, const int* in_sizes, int n_in,
                              void* d_out, int out_size, void* d_ws, size_t ws_size,
                              hipStream_t stream) {
  const float* x      = (const float*)d_in[0];
  const float* enc_w1 = (const float*)d_in[1];
  const float* enc_b1 = (const float*)d_in[2];
  const float* enc_w2 = (const float*)d_in[3];
  const float* enc_b2 = (const float*)d_in[4];
  const float* conv_w = (const float*)d_in[5];
  const float* conv_b = (const float*)d_in[6];
  const float* cbook  = (const float*)d_in[7];
  const float* dec_w1 = (const float*)d_in[8];
  const float* dec_b1 = (const float*)d_in[9];
  const float* dec_w2 = (const float*)d_in[10];
  const float* dec_b2 = (const float*)d_in[11];

  char* ws = (char*)d_ws;
  size_t off = 0;
  auto alloc = [&](size_t bytes) -> void* {
    off = (off + 255) & ~(size_t)255;
    void* p = ws + off;
    off += bytes;
    return p;
  };
  bf16_t* xb    = (bf16_t*)alloc((size_t)BB * INN * 2);
  bf16_t* w1t   = (bf16_t*)alloc((size_t)INN * HH1 * 2);   // [H1, IN]
  bf16_t* w2t   = (bf16_t*)alloc((size_t)HH1 * HH2 * 2);   // [H2, H1]
  bf16_t* wd1t  = (bf16_t*)alloc((size_t)DK * HH3 * 2);    // [H3, DK]
  bf16_t* wd2t  = (bf16_t*)alloc((size_t)HH3 * HH4 * 2);   // [H4, H3]
  bf16_t* cbb   = (bf16_t*)alloc((size_t)KC * DD * 2);
  float*  cnorm = (float*)alloc((size_t)KC * 4);
  bf16_t* h     = (bf16_t*)alloc((size_t)BB * HH1 * 2);
  float*  z     = (float*)alloc((size_t)BB * HH2 * 4);
  bf16_t* flatb = (bf16_t*)alloc((size_t)M2 * DD * 2);
  int*    vidx  = (int*)alloc((size_t)M2 * 4);
  bf16_t* qb    = (bf16_t*)alloc((size_t)BB * DK * 2);
  bf16_t* hq    = (bf16_t*)alloc((size_t)BB * HH3 * 2);
  float*  part  = (float*)alloc((size_t)(M2 / 8) * 4);
  int*    hist  = (int*)alloc((size_t)KC * 4);
  (void)ws_size; (void)in_sizes; (void)n_in; (void)out_size;

  float* out    = (float*)d_out;
  float* o_loss = out;
  float* o_xrec = out + 1;                            // [B, 512]
  float* o_perp = out + 1 + (size_t)BB * HH4;
  float* o_quan = out + 2 + (size_t)BB * HH4;         // [B, D, H2]

  // prep
  zero_hist_kernel<<<1, 256, 0, stream>>>(hist);
  cvt_bf16_kernel<<<(BB * INN) / 256, 256, 0, stream>>>(x, xb, BB * INN);
  transpose_bf16_kernel<<<(INN * HH1) / 256, 256, 0, stream>>>(enc_w1, w1t, INN, HH1);
  transpose_bf16_kernel<<<(HH1 * HH2) / 256, 256, 0, stream>>>(enc_w2, w2t, HH1, HH2);
  transpose_bf16_kernel<<<(DK * HH3) / 256, 256, 0, stream>>>(dec_w1, wd1t, DK, HH3);
  transpose_bf16_kernel<<<(HH3 * HH4) / 256, 256, 0, stream>>>(dec_w2, wd2t, HH3, HH4);
  codebook_prep_kernel<<<1, 256, 0, stream>>>(cbook, cbb, cnorm);

  // encoder
  gemm_bf16_kernel<<<gemm_blocks(BB, HH1), 256, 0, stream>>>(
      xb, w1t, enc_b1, nullptr, h, BB, HH1, INN, 1);
  gemm_bf16_kernel<<<gemm_blocks(BB, HH2), 256, 0, stream>>>(
      h, w2t, enc_b2, z, nullptr, BB, HH2, HH1, 1);

  // pre-VQ conv + VQ
  conv_prevq_kernel<<<(int)(((size_t)M2 * DD) / 256), 256, 0, stream>>>(z, conv_w, conv_b, flatb);
  vq_argmin_kernel<<<(M2 / 16 + 7) / 8, 256, 0, stream>>>(flatb, cbb, cnorm, vidx);
  quantize_kernel<<<M2 / 8, 256, 0, stream>>>(vidx, cbook, z, conv_w, conv_b,
                                              o_quan, qb, part, hist);

  // decoder
  gemm_bf16_kernel<<<gemm_blocks(BB, HH3), 256, 0, stream>>>(
      qb, wd1t, dec_b1, nullptr, hq, BB, HH3, DK, 1);
  gemm_bf16_kernel<<<gemm_blocks(BB, HH4), 256, 0, stream>>>(
      hq, wd2t, dec_b2, o_xrec, nullptr, BB, HH4, HH3, 0);

  // scalars
  finalize_kernel<<<1, 256, 0, stream>>>(part, M2 / 8, hist, o_loss, o_perp);
}
